// DeepARPoint_7919919694316
// MI455X (gfx1250) — matrix operand order repaired and checked by reading, not compiled, not run on hardware
//
#include <hip/hip_runtime.h>
#include <cstddef>
#include <cstdint>

// ---------------------------------------------------------------------------
// DeepAR 2-layer LSTM (B=512, HID=512, 168 enc + 23 dec steps) for gfx1250.
// One workgroup per batch tile of 16 (no inter-WG sync), 8 waves per WG
// (4 waves/SIMD in CU mode -> ~256 VGPR budget, no accumulator spills).
// Matmuls in fp16 via v_wmma_f32_16x16x32_f16; weights pre-packed into the
// wave32 B-fragment layout so each lane reads one contiguous 32B chunk via
// global_load_b128.  An opaque zero SGPR ("jitter") re-materialized per time
// step keeps weight/bias loads inside the loop (L2 streaming, no LICM spill).
// LSTM nonlinearities use the gfx1250 hardware v_tanh_f32 transcendental.
// ---------------------------------------------------------------------------

typedef __attribute__((ext_vector_type(16))) _Float16 v16h;
typedef __attribute__((ext_vector_type(8)))  float    v8f;
typedef __attribute__((ext_vector_type(4)))  float    f32x4;

#define CTX      168
#define TSTEPS   191      // 168 encoder + 23 decoder
#define HID      512
#define LSEQ     219      // MAX_LAG + CTX + H - 1
#define PASTN    196      // L - (H-1)
#define BATCH    512
#define BT       16
#define NBLK     (BATCH / BT)   // 32 workgroups

// d_ws byte offsets (fp16 weight fragments + prep buffers), all 256B aligned
#define OFF_IH0     0u          //  128*1*512 halves  = 128KB  (K padded 10->32)
#define OFF_HH0     131072u     //  128*16*512 halves = 2MB
#define OFF_IH1     2228224u    //  2MB
#define OFF_HH1     4325376u    //  2MB
#define OFF_ENC     6422528u    //  512*168*10 f32
#define OFF_COV     9863168u    //  512*23*3  f32
#define OFF_PAST    10004480u   //  512*196   f32
#define OFF_SCALE   10405888u   //  512 f32
#define OFF_LSCALE  10407936u   //  512 f32

// gfx1250 hardware tanh (TRANS32) with portable fallback
__device__ __forceinline__ float fast_tanh(float x) {
#if __has_builtin(__builtin_amdgcn_tanhf)
  return __builtin_amdgcn_tanhf(x);
#elif __has_builtin(__builtin_amdgcn_tanh_f32)
  return __builtin_amdgcn_tanh_f32(x);
#else
  return tanhf(x);
#endif
}
// sigmoid via tanh: 1 transcendental + 2 VALU
__device__ __forceinline__ float sigm(float x) {
  return 0.5f * fast_tanh(0.5f * x) + 0.5f;
}

__device__ __forceinline__ v8f splat8(float v) {
  v8f r;
#pragma unroll
  for (int i = 0; i < 8; ++i) r[i] = v;
  return r;
}

// B fragment: lane's 16 halves are contiguous in the packed buffer (32B)
__device__ __forceinline__ v16h ldB(const _Float16* __restrict__ p) {
  v16h r;
  ((f32x4*)&r)[0] = ((const f32x4*)p)[0];
  ((f32x4*)&r)[1] = ((const f32x4*)p)[1];
  return r;
}

// A fragment (16x32 f16) from an LDS activation buffer.
// Lane l holds row M = l&15; K-halves {kb..kb+7, kb+16..kb+23} with
// kb = kt*32 + (l<16 ? 0 : 8)  -> two 16B LDS loads.
__device__ __forceinline__ v16h ldA(const _Float16* row, int kcol0) {
  v16h r;
  ((f32x4*)&r)[0] = *(const f32x4*)(row + kcol0);
  ((f32x4*)&r)[1] = *(const f32x4*)(row + kcol0 + 16);
  return r;
}

#define WMMA_F16(a, b, c) \
  __builtin_amdgcn_wmma_f32_16x16x32_f16(false, (a), false, (b), (short)0, (c), false, false)

// ------------------------------ prep kernels -------------------------------

__global__ void k_scale(const float* __restrict__ X, float* __restrict__ scale,
                        float* __restrict__ logscale) {
  __shared__ float red[256];
  int b = blockIdx.x;
  float s = 0.f;
  for (int t = threadIdx.x; t < CTX; t += 256)
    s += fabsf(X[((size_t)b * LSEQ + 28 + t) * 3]);
  red[threadIdx.x] = s;
  __syncthreads();
  for (int k = 128; k > 0; k >>= 1) {
    if (threadIdx.x < k) red[threadIdx.x] += red[threadIdx.x + k];
    __syncthreads();
  }
  if (threadIdx.x == 0) {
    float sc = fmaxf(red[0] / (float)CTX, 0.001f);
    scale[b] = sc;
    logscale[b] = logf(sc);
  }
}

__global__ void k_past(const float* __restrict__ X, const float* __restrict__ scale,
                       float* __restrict__ past) {
  int i = blockIdx.x * 256 + threadIdx.x;
  if (i >= BATCH * PASTN) return;
  int j = i % PASTN, b = i / PASTN;
  past[i] = X[((size_t)b * LSEQ + (PASTN - 1 - j)) * 3] / scale[b];
}

__global__ void k_encin(const float* __restrict__ X, const float* __restrict__ past,
                        const float* __restrict__ scale, const float* __restrict__ logscale,
                        float* __restrict__ enc_in) {
  int i = blockIdx.x * 256 + threadIdx.x;
  if (i >= BATCH * CTX * 10) return;
  int k = i % 10, t = (i / 10) % CTX, b = i / (CTX * 10);
  float v;
  if (k == 0)       v = X[((size_t)b * LSEQ + 28 + t) * 3] / scale[b];
  else if (k < 3)   v = X[((size_t)b * LSEQ + 28 + t) * 3 + k];
  else if (k == 3)  v = logscale[b];
  else {
    const int LAGS[6] = {1, 2, 3, 7, 14, 28};
    v = past[(size_t)b * PASTN + (CTX - 1 - t) + LAGS[k - 4]];
  }
  enc_in[i] = v;
}

__global__ void k_covs(const float* __restrict__ X, const float* __restrict__ logscale,
                       float* __restrict__ covs) {
  int i = blockIdx.x * 256 + threadIdx.x;
  if (i >= BATCH * 23 * 3) return;
  int c = i % 3, d = (i / 3) % 23, b = i / 69;
  covs[i] = (c == 2) ? logscale[b]
                     : X[((size_t)b * LSEQ + PASTN + d) * 3 + 1 + c];
}

// Pack W (2048 x Kdim, row-major f32) into wave32 B-fragment layout, fp16.
// Fragment (ntile, kt): lane<16 -> col N=lane, K=kt*32+{0..7,16..23};
//                       lane>=16 -> col N=lane-16, K=kt*32+{8..15,24..31}.
__global__ void k_pack(const float* __restrict__ W, _Float16* __restrict__ out,
                       int Kdim, int NKT) {
  int o = blockIdx.x * 256 + threadIdx.x;
  int total = 128 * NKT * 512;
  if (o >= total) return;
  int idx  = o & 15;
  int lane = (o >> 4) & 31;
  int kt   = (o >> 9) % NKT;
  int nt   = o / (512 * NKT);
  int n  = lane & 15;
  int kb = (lane < 16) ? 0 : 8;
  int k  = kt * 32 + kb + (idx & 7) + ((idx >> 3) << 4);
  int row = nt * 16 + n;
  float v = (k < Kdim) ? W[(size_t)row * Kdim + k] : 0.0f;
  out[o] = (_Float16)v;
}

// ------------------------------ main LSTM kernel ---------------------------

__global__ void __launch_bounds__(256)
__attribute__((amdgpu_waves_per_eu(4, 4)))
deepar_main(const float* __restrict__ enc_in,
            const float* __restrict__ covs,
            const float* __restrict__ past,
            const float* __restrict__ scale,
            const _Float16* __restrict__ gW_ih0,
            const _Float16* __restrict__ gW_hh0,
            const _Float16* __restrict__ gW_ih1,
            const _Float16* __restrict__ gW_hh1,
            const float* __restrict__ b0,
            const float* __restrict__ b1,
            const float* __restrict__ W_head,
            const float* __restrict__ b_head,
            float* __restrict__ out) {
  __shared__ __align__(16) _Float16 h1buf[16][HID];
  __shared__ __align__(16) _Float16 h2buf[16][HID];
  __shared__ __align__(16) _Float16 xin[16][32];
  __shared__ float ring[16][32];
  __shared__ float part[16][16];
  __shared__ float scl[16];
  __shared__ float wh[HID];

  const int tid  = threadIdx.x;
  const int w    = tid >> 5;        // wave 0..7
  const int lane = tid & 31;
  const int nlo  = lane & 15;
  const int klo  = (lane < 16) ? 0 : 8;
  const int mrow = ((lane >= 16) ? 8 : 0);
  const int b0i  = blockIdx.x * BT;

  // zero LDS activations; stage W_head/scale into LDS; load x_0
  for (int i = tid; i < 16 * HID; i += 256) {
    ((_Float16*)h1buf)[i] = (_Float16)0.f;
    ((_Float16*)h2buf)[i] = (_Float16)0.f;
  }
  for (int i = tid; i < 16 * 32; i += 256) ((_Float16*)xin)[i] = (_Float16)0.f;
  for (int i = tid; i < HID; i += 256) wh[i] = W_head[i];
  if (tid < 16) scl[tid] = scale[b0i + tid];
  __syncthreads();
  if (tid < 160) {
    int m = tid / 10, k = tid % 10;
    xin[m][k] = (_Float16)enc_in[((size_t)(b0i + m) * CTX + 0) * 10 + k];
  }
  __syncthreads();

  // Single per-lane voffsets (halves); (g,s) selection via immediate offsets.
  const int voffN16 = w * 32768 + lane * 16;  // ntile = g*32 + w*4 + s, NKT=16
  const int voffN1  = w * 2048  + lane * 16;  // W_ih0, NKT=1
  const float bhead = b_head[0];

  v8f c1[4], c2[4];
#pragma unroll
  for (int s = 0; s < 4; ++s) { c1[s] = splat8(0.f); c2[s] = splat8(0.f); }
  v8f acc[4][4];

  for (int t = 0; t < TSTEPS; ++t) {
    // Anti-LICM: opaque zero SGPR keeps weight/bias loads loop-variant.
    unsigned jitter = 0;
    asm volatile("" : "+s"(jitter));

    // ============================ layer 0 ================================
#pragma unroll
    for (int g = 0; g < 4; ++g)
#pragma unroll
      for (int s = 0; s < 4; ++s)
        acc[g][s] = splat8(b0[g * 512 + w * 64 + s * 16 + nlo + jitter]);
    {  // input contribution (K padded 10 -> 32, single k-tile)
      v16h ax = ldA(&xin[nlo][0], klo);
#pragma unroll
      for (int g = 0; g < 4; ++g)
#pragma unroll
        for (int s = 0; s < 4; ++s) {
          v16h bf = ldB(gW_ih0 + voffN1 + jitter + (g * 16384 + s * 512));
          acc[g][s] = WMMA_F16(ax, bf, acc[g][s]);
        }
    }
#pragma unroll 1
    for (int kt = 0; kt < 16; ++kt) {  // recurrent contribution
      v16h ah = ldA(&h1buf[nlo][0], kt * 32 + klo);
#pragma unroll
      for (int g = 0; g < 4; ++g)
#pragma unroll
        for (int s = 0; s < 4; ++s) {
          v16h bf = ldB(gW_hh0 + voffN16 + jitter + (g * 262144 + s * 8192) + kt * 512);
          acc[g][s] = WMMA_F16(ah, bf, acc[g][s]);
        }
    }
    __syncthreads();   // everyone done reading h1_prev
#pragma unroll
    for (int s = 0; s < 4; ++s)
#pragma unroll
      for (int r = 0; r < 8; ++r) {
        float iv = sigm(acc[0][s][r]);
        float fv = sigm(acc[1][s][r]);
        float gv = fast_tanh(acc[2][s][r]);
        float ov = sigm(acc[3][s][r]);
        float cn = fv * c1[s][r] + iv * gv;
        c1[s][r] = cn;
        h1buf[r + mrow][w * 64 + s * 16 + nlo] = (_Float16)(ov * fast_tanh(cn));
      }
    __syncthreads();

    // ============================ layer 1 ================================
#pragma unroll
    for (int g = 0; g < 4; ++g)
#pragma unroll
      for (int s = 0; s < 4; ++s)
        acc[g][s] = splat8(b1[g * 512 + w * 64 + s * 16 + nlo + jitter]);
#pragma unroll 1
    for (int kt = 0; kt < 16; ++kt) {
      v16h a1 = ldA(&h1buf[nlo][0], kt * 32 + klo);
#pragma unroll
      for (int g = 0; g < 4; ++g)
#pragma unroll
        for (int s = 0; s < 4; ++s) {
          v16h bf = ldB(gW_ih1 + voffN16 + jitter + (g * 262144 + s * 8192) + kt * 512);
          acc[g][s] = WMMA_F16(a1, bf, acc[g][s]);
        }
      v16h a2 = ldA(&h2buf[nlo][0], kt * 32 + klo);
#pragma unroll
      for (int g = 0; g < 4; ++g)
#pragma unroll
        for (int s = 0; s < 4; ++s) {
          v16h bf = ldB(gW_hh1 + voffN16 + jitter + (g * 262144 + s * 8192) + kt * 512);
          acc[g][s] = WMMA_F16(a2, bf, acc[g][s]);
        }
    }
    __syncthreads();   // everyone done reading h2_prev
#pragma unroll
    for (int s = 0; s < 4; ++s)
#pragma unroll
      for (int r = 0; r < 8; ++r) {
        float iv = sigm(acc[0][s][r]);
        float fv = sigm(acc[1][s][r]);
        float gv = fast_tanh(acc[2][s][r]);
        float ov = sigm(acc[3][s][r]);
        float cn = fv * c2[s][r] + iv * gv;
        c2[s][r] = cn;
        h2buf[r + mrow][w * 64 + s * 16 + nlo] = (_Float16)(ov * fast_tanh(cn));
      }
    __syncthreads();

    // ============================== head =================================
    {
      int m = tid >> 4, seg = tid & 15;
      float p = 0.f;
#pragma unroll 8
      for (int k = 0; k < 32; ++k)
        p += (float)h2buf[m][seg * 32 + k] * wh[seg * 32 + k];
      part[m][seg] = p;
    }
    __syncthreads();

    // ================= head reduce + output + next input ================
    {
      int tn = t + 1;
      if (tid < 16) {
        float y = bhead;
#pragma unroll
        for (int s2 = 0; s2 < 16; ++s2) y += part[tid][s2];
        out[(size_t)(b0i + tid) * TSTEPS + t] = y * scl[tid];
        if (tn >= CTX && tn < TSTEPS) {    // decoder input for step tn
          int d = tn - CTX;                // 0..22
          int m = tid;
          if (d == 0)
            for (int j = 0; j < 29; ++j)
              ring[m][j] = past[(size_t)(b0i + m) * PASTN + j];
          float pred = y;                  // thread-local feedback
          int head = (32 - (d + 1)) & 31;  // ring head after (d+1) pushes
          ring[m][head] = pred;
          xin[m][0] = (_Float16)pred;
          const float* cv = covs + ((size_t)(b0i + m) * 23 + d) * 3;
          xin[m][1] = (_Float16)cv[0];
          xin[m][2] = (_Float16)cv[1];
          xin[m][3] = (_Float16)cv[2];
          const int LAGS[6] = {1, 2, 3, 7, 14, 28};
#pragma unroll
          for (int j = 0; j < 6; ++j)
            xin[m][4 + j] = (_Float16)ring[m][(head + LAGS[j]) & 31];
        }
      }
      if (tn < CTX && tid < 160) {         // encoder input for step tn
        int m = tid / 10, k = tid % 10;
        xin[m][k] = (_Float16)enc_in[((size_t)(b0i + m) * CTX + tn) * 10 + k];
      }
    }
    __syncthreads();
  }
}

// ------------------------------ launch -------------------------------------

extern "C" void kernel_launch(void* const* d_in, const int* in_sizes, int n_in,
                              void* d_out, int out_size, void* d_ws, size_t ws_size,
                              hipStream_t stream) {
  (void)in_sizes; (void)n_in; (void)out_size; (void)ws_size;
  const float* X      = (const float*)d_in[0];
  // d_in[1] = pad_mask (all ones), d_in[2] = H, d_in[3] = context_length: constants here
  const float* W_ih0  = (const float*)d_in[4];
  const float* W_hh0  = (const float*)d_in[5];
  const float* b0     = (const float*)d_in[6];
  const float* W_ih1  = (const float*)d_in[7];
  const float* W_hh1  = (const float*)d_in[8];
  const float* b1     = (const float*)d_in[9];
  const float* W_head = (const float*)d_in[10];
  const float* b_head = (const float*)d_in[11];
  float* out = (float*)d_out;

  char* ws = (char*)d_ws;
  _Float16* fW_ih0 = (_Float16*)(ws + OFF_IH0);
  _Float16* fW_hh0 = (_Float16*)(ws + OFF_HH0);
  _Float16* fW_ih1 = (_Float16*)(ws + OFF_IH1);
  _Float16* fW_hh1 = (_Float16*)(ws + OFF_HH1);
  float* enc_in   = (float*)(ws + OFF_ENC);
  float* covsb    = (float*)(ws + OFF_COV);
  float* pastb    = (float*)(ws + OFF_PAST);
  float* scaleb   = (float*)(ws + OFF_SCALE);
  float* lscaleb  = (float*)(ws + OFF_LSCALE);

  k_scale<<<BATCH, 256, 0, stream>>>(X, scaleb, lscaleb);
  k_past<<<(BATCH * PASTN + 255) / 256, 256, 0, stream>>>(X, scaleb, pastb);
  k_encin<<<(BATCH * CTX * 10 + 255) / 256, 256, 0, stream>>>(X, pastb, scaleb, lscaleb, enc_in);
  k_covs<<<(BATCH * 23 * 3 + 255) / 256, 256, 0, stream>>>(X, lscaleb, covsb);
  k_pack<<<(128 * 1 * 512 + 255) / 256, 256, 0, stream>>>(W_ih0, fW_ih0, 10, 1);
  k_pack<<<(128 * 16 * 512 + 255) / 256, 256, 0, stream>>>(W_hh0, fW_hh0, 512, 16);
  k_pack<<<(128 * 16 * 512 + 255) / 256, 256, 0, stream>>>(W_ih1, fW_ih1, 512, 16);
  k_pack<<<(128 * 16 * 512 + 255) / 256, 256, 0, stream>>>(W_hh1, fW_hh1, 512, 16);

  deepar_main<<<NBLK, 256, 0, stream>>>(enc_in, covsb, pastb, scaleb,
                                        fW_ih0, fW_hh0, fW_ih1, fW_hh1,
                                        b0, b1, W_head, b_head, out);
}
